// CongestionWrapperEncoder0_40089224741034
// MI455X (gfx1250) — matrix operand order repaired and checked by reading, not compiled, not run on hardware
//
#include <hip/hip_runtime.h>
#include <hip/hip_bf16.h>
#include <hip/hip_fp16.h>
#include <stdint.h>

// Problem constants (from reference)
#define G_    16        // B*DAYS
#define N_    10000
#define E_    80000
#define ET_   90000     // E + N self loops
#define CIN_  32
#define H_    4
#define HC_   128       // H * C_OUT
#define NTILE_ 625      // N_/16 node tiles per graph
#define NEG_SLOPE_ 0.2f

typedef __attribute__((ext_vector_type(16))) _Float16 v16h;
typedef __attribute__((ext_vector_type(8)))  float    v8f;

__device__ __forceinline__ unsigned pack2_f16(float a, float b) {
  _Float16 la = (_Float16)a, lb = (_Float16)b;
  unsigned short ua = __builtin_bit_cast(unsigned short, la);
  unsigned short ub = __builtin_bit_cast(unsigned short, lb);
  return (unsigned)ua | ((unsigned)ub << 16);
}

// ---------------------------------------------------------------------------
// Kernel 0: pack W [32,128] f32 into the WMMA B-operand register image, f16.
// Linear layout: entry (j*32 + lane) holds that lane's 8 dwords (16 f16) for
// column tile j:  dword v = { K = (v<4 ? 8*grp + 2v : 16 + 8*grp + 2(v-4)), +1 } at column j*16 + (lane&15).
// 8 KB total; done once, then broadcast to every GEMM block via LDS.
// ---------------------------------------------------------------------------
__global__ __launch_bounds__(256) void k_wprep(const float* __restrict__ W,
                                               unsigned* __restrict__ Wpk)
{
  const int tid  = threadIdx.x;      // 0..255 : j = tid>>5, lane = tid&31
  const int lane = tid & 31;
  const int j    = tid >> 5;
  const int m    = lane & 15;
  const int grp  = lane >> 4;
  const int col  = j * 16 + m;
  #pragma unroll
  for (int v = 0; v < 8; ++v) {
    const int k0 = (v < 4) ? (grp * 8 + 2 * v) : (16 + grp * 8 + 2 * (v - 4));
    Wpk[tid * 8 + v] = pack2_f16(W[(size_t)k0 * HC_ + col],
                                 W[(size_t)(k0 + 1) * HC_ + col]);
  }
}

// ---------------------------------------------------------------------------
// Kernel 1: xh[g,n,:] = (embedding[x[g,n]]) @ W  via v_wmma_f32_16x16x32_f16.
// 128 threads = 4 waves per block; packed W staged into LDS with
// global_load_async_to_lds_b128 (ASYNCcnt), each wave owns one 16-node tile.
// ---------------------------------------------------------------------------
__global__ __launch_bounds__(128) void k_gemm_wmma(
    const int* __restrict__ x, const float* __restrict__ emb,
    const unsigned* __restrict__ Wpk, float* __restrict__ xh)
{
  __shared__ __align__(32) unsigned lds_w[2048];   // 8 KB packed-f16 W image

  const int tid  = threadIdx.x;
  const int wave = tid >> 5;
  const int lane = tid & 31;
  const int m    = lane & 15;          // row within tile / col within D tile
  const int grp  = lane >> 4;          // lane half
  const int g    = blockIdx.y;

  // ---- async-stage the 8 KB packed W into LDS (CDNA5 ASYNCcnt path) ------
  {
    unsigned lofs = (unsigned)(uintptr_t)(void*)lds_w + (unsigned)tid * 16u;
    unsigned long long ga =
        (unsigned long long)(uintptr_t)Wpk + (unsigned long long)tid * 16u;
    #pragma unroll
    for (int q = 0; q < 4; ++q) {      // 128 thr * 16 B * 4 = 8192 B
      asm volatile("global_load_async_to_lds_b128 %0, %1, off"
                   :: "v"(lofs), "v"(ga) : "memory");
      lofs += 2048u;
      ga   += 2048ull;
    }
#if __has_builtin(__builtin_amdgcn_s_wait_asynccnt)
    __builtin_amdgcn_s_wait_asynccnt(0);
#else
    asm volatile("s_wait_asynccnt 0" ::: "memory");
#endif
  }
  __syncthreads();                     // all waves see the staged W

  const int tile = blockIdx.x * 4 + wave;   // 16-node tile of this wave
  if (tile >= NTILE_) return;

  // ---- A matrix: 16x32 f16, ISA layout:
  //  lanes 0-15 : K = 0..7 (v0-3), 16..23 (v4-7)
  //  lanes16-31 : K = 8..15(v0-3), 24..31 (v4-7)
  const int node = tile * 16 + m;
  const int idx  = x[(size_t)g * N_ + node];
  const float* erow = emb + (size_t)idx * CIN_;

  const float4* c0 = reinterpret_cast<const float4*>(erow + grp * 8);
  const float4* c1 = reinterpret_cast<const float4*>(erow + 16 + grp * 8);
  float4 f0 = c0[0], f1 = c0[1], f2 = c1[0], f3 = c1[1];

  v16h a;
  a[0]=(_Float16)f0.x;  a[1]=(_Float16)f0.y;  a[2]=(_Float16)f0.z;  a[3]=(_Float16)f0.w;
  a[4]=(_Float16)f1.x;  a[5]=(_Float16)f1.y;  a[6]=(_Float16)f1.z;  a[7]=(_Float16)f1.w;
  a[8]=(_Float16)f2.x;  a[9]=(_Float16)f2.y;  a[10]=(_Float16)f2.z; a[11]=(_Float16)f2.w;
  a[12]=(_Float16)f3.x; a[13]=(_Float16)f3.y; a[14]=(_Float16)f3.z; a[15]=(_Float16)f3.w;

  float* orow = xh + ((size_t)g * N_ + (size_t)tile * 16) * HC_;

  #pragma unroll
  for (int j = 0; j < 8; ++j) {
    const int col = j * 16 + m;        // global output column of this lane
    // B fragment: 16 f16 (32 B) per lane straight from LDS (2x ds_load_b128)
    const v16h b =
        *reinterpret_cast<const v16h*>(&lds_w[(size_t)(j * 32 + lane) * 8]);
    v8f acc = {};
    acc = __builtin_amdgcn_wmma_f32_16x16x32_f16(
        /*neg_a=*/false, a, /*neg_b=*/false, b,
        /*c_mod=*/(short)0, acc, /*reuse_a=*/false, /*reuse_b=*/false);
    // D layout: VGPR r -> row (r + 8*grp), lane -> column col
    #pragma unroll
    for (int r = 0; r < 8; ++r) {
      const int row = r + 8 * grp;
      orow[(size_t)row * HC_ + col] = acc[r];
    }
  }
}

// ---------------------------------------------------------------------------
// Kernel 2: per-node attention logits a_src/a_dst  [G*N, H]
// ---------------------------------------------------------------------------
__global__ void k_att(const float* __restrict__ xh,
                      const float* __restrict__ att_src,
                      const float* __restrict__ att_dst,
                      float* __restrict__ asrc, float* __restrict__ adst)
{
  const long long t = (long long)blockIdx.x * blockDim.x + threadIdx.x; // g*N+n
  if (t >= (long long)G_ * N_) return;
  const float* row = xh + (size_t)t * HC_;
  #pragma unroll
  for (int h = 0; h < H_; ++h) {
    const float4* r4 = reinterpret_cast<const float4*>(row + h * 32);
    const float4* s4 = reinterpret_cast<const float4*>(att_src + h * 32);
    const float4* d4 = reinterpret_cast<const float4*>(att_dst + h * 32);
    float ss = 0.f, dd = 0.f;
    #pragma unroll
    for (int q = 0; q < 8; ++q) {
      const float4 v = r4[q], s = s4[q], d = d4[q];
      ss += v.x * s.x + v.y * s.y + v.z * s.z + v.w * s.w;
      dd += v.x * d.x + v.y * d.y + v.z * d.z + v.w * d.w;
    }
    asrc[(size_t)t * H_ + h] = ss;
    adst[(size_t)t * H_ + h] = dd;
  }
}

// ---------------------------------------------------------------------------
// Kernel 3: init out with bias, m with encoded -inf, s with 0
// ---------------------------------------------------------------------------
__global__ void k_init(float* __restrict__ out, const float* __restrict__ bias,
                       unsigned* __restrict__ mbuf, float* __restrict__ sbuf)
{
  const long long i = (long long)blockIdx.x * blockDim.x + threadIdx.x;
  if (i < (long long)G_ * N_ * HC_) out[i] = bias[i & (HC_ - 1)];
  if (i < (long long)G_ * N_ * H_) {
    mbuf[i] = 0x007FFFFFu;  // order-preserving encoding of -inf (~bits(-inf))
    sbuf[i] = 0.f;
  }
}

__device__ __forceinline__ void edge_nodes(const int* __restrict__ adj, int e,
                                           int& s, int& d)
{
  if (e < E_) { s = adj[e]; d = adj[E_ + e]; }
  else        { s = e - E_; d = e - E_; }    // self loop
}

__device__ __forceinline__ unsigned ord_encode(float f) {
  unsigned u = __float_as_uint(f);
  return (u & 0x80000000u) ? ~u : (u | 0x80000000u);
}
__device__ __forceinline__ float ord_decode(unsigned u) {
  u = (u & 0x80000000u) ? (u ^ 0x80000000u) : ~u;
  return __uint_as_float(u);
}

// ---------------------------------------------------------------------------
// Kernel 4: segment max of leaky-relu logits (atomic max on ordered uint)
// ---------------------------------------------------------------------------
__global__ void k_edge_max(const int* __restrict__ adj,
                           const float* __restrict__ asrc,
                           const float* __restrict__ adst,
                           unsigned* __restrict__ mbuf)
{
  const long long t = (long long)blockIdx.x * blockDim.x + threadIdx.x;
  if (t >= (long long)G_ * ET_) return;
  const int g = (int)(t / ET_), e = (int)(t % ET_);
  int s, d; edge_nodes(adj, e, s, d);
  const float* ps = asrc + ((size_t)g * N_ + s) * H_;
  const float* pd = adst + ((size_t)g * N_ + d) * H_;
  unsigned* pm = mbuf + ((size_t)g * N_ + d) * H_;
  #pragma unroll
  for (int h = 0; h < H_; ++h) {
    float al = ps[h] + pd[h];
    al = (al >= 0.f) ? al : NEG_SLOPE_ * al;
    atomicMax(&pm[h], ord_encode(al));
  }
}

// ---------------------------------------------------------------------------
// Kernel 5: segment sum of exp(alpha - m)
// ---------------------------------------------------------------------------
__global__ void k_edge_sum(const int* __restrict__ adj,
                           const float* __restrict__ asrc,
                           const float* __restrict__ adst,
                           const unsigned* __restrict__ mbuf,
                           float* __restrict__ sbuf)
{
  const long long t = (long long)blockIdx.x * blockDim.x + threadIdx.x;
  if (t >= (long long)G_ * ET_) return;
  const int g = (int)(t / ET_), e = (int)(t % ET_);
  int s, d; edge_nodes(adj, e, s, d);
  const float* ps = asrc + ((size_t)g * N_ + s) * H_;
  const float* pd = adst + ((size_t)g * N_ + d) * H_;
  const unsigned* pm = mbuf + ((size_t)g * N_ + d) * H_;
  float* pss = sbuf + ((size_t)g * N_ + d) * H_;
  #pragma unroll
  for (int h = 0; h < H_; ++h) {
    float al = ps[h] + pd[h];
    al = (al >= 0.f) ? al : NEG_SLOPE_ * al;
    const float ev = __expf(al - ord_decode(pm[h]));
    atomicAdd(&pss[h], ev);
  }
}

// ---------------------------------------------------------------------------
// Kernel 6: weighted scatter-add: out[g,dst,col] += w(g,e,h) * xh[g,src,col].
// 128 consecutive threads cover the 128 channels of one (g,e): loads and
// float atomics are fully coalesced 512B transactions resolving in L2
// (xh + out working set ~164 MB fits the 192 MB L2).
// ---------------------------------------------------------------------------
__global__ void k_edge_agg(const int* __restrict__ adj,
                           const float* __restrict__ asrc,
                           const float* __restrict__ adst,
                           const unsigned* __restrict__ mbuf,
                           const float* __restrict__ sbuf,
                           const float* __restrict__ xh,
                           float* __restrict__ out)
{
  const long long t = (long long)blockIdx.x * blockDim.x + threadIdx.x;
  if (t >= (long long)G_ * ET_ * HC_) return;
  const int col = (int)(t & (HC_ - 1));
  const int h   = col >> 5;
  const long long ge = t >> 7;              // (g, e)
  const int g = (int)(ge / ET_), e = (int)(ge % ET_);
  int s, d; edge_nodes(adj, e, s, d);

  float al = asrc[((size_t)g * N_ + s) * H_ + h] +
             adst[((size_t)g * N_ + d) * H_ + h];
  al = (al >= 0.f) ? al : NEG_SLOPE_ * al;
  const float mval = ord_decode(mbuf[((size_t)g * N_ + d) * H_ + h]);
  const float ev   = __expf(al - mval);
  const float w    = ev / (sbuf[((size_t)g * N_ + d) * H_ + h] + 1e-16f);

  const float v = xh[((size_t)g * N_ + s) * HC_ + col];
  atomicAdd(&out[((size_t)g * N_ + d) * HC_ + col], w * v);
}

// ---------------------------------------------------------------------------
extern "C" void kernel_launch(void* const* d_in, const int* in_sizes, int n_in,
                              void* d_out, int out_size, void* d_ws, size_t ws_size,
                              hipStream_t stream)
{
  const int*   x       = (const int*)  d_in[0];   // [2,8,10000] int32
  const int*   adj     = (const int*)  d_in[1];   // [2,80000]   int32
  const float* emb     = (const float*)d_in[2];   // [10000,32]
  const float* W       = (const float*)d_in[3];   // [32,128]
  const float* att_src = (const float*)d_in[4];   // [4,32]
  const float* att_dst = (const float*)d_in[5];   // [4,32]
  const float* bias    = (const float*)d_in[6];   // [128]
  float* out = (float*)d_out;                     // [2,8,10000,128]

  char* ws = (char*)d_ws;
  size_t off = 0;
  float* xh = (float*)(ws + off);     off += (size_t)G_ * N_ * HC_ * sizeof(float);
  float* asrc = (float*)(ws + off);   off += (size_t)G_ * N_ * H_ * sizeof(float);
  float* adst = (float*)(ws + off);   off += (size_t)G_ * N_ * H_ * sizeof(float);
  unsigned* mbuf = (unsigned*)(ws + off); off += (size_t)G_ * N_ * H_ * sizeof(unsigned);
  float* sbuf = (float*)(ws + off);   off += (size_t)G_ * N_ * H_ * sizeof(float);
  unsigned* Wpk = (unsigned*)(ws + off); off += 2048 * sizeof(unsigned);

  // 0) pack W into WMMA B-operand f16 image (once)
  k_wprep<<<1, 256, 0, stream>>>(W, Wpk);

  // 1) GEMM via WMMA: 4 waves/block, one 16-node tile per wave
  k_gemm_wmma<<<dim3((NTILE_ + 3) / 4, G_), 128, 0, stream>>>(x, emb, Wpk, xh);

  // 2) attention logits
  {
    const long long n = (long long)G_ * N_;
    k_att<<<(unsigned)((n + 255) / 256), 256, 0, stream>>>(xh, att_src, att_dst, asrc, adst);
  }
  // 3) init out/m/s
  {
    const long long n = (long long)G_ * N_ * HC_;
    k_init<<<(unsigned)((n + 255) / 256), 256, 0, stream>>>(out, bias, mbuf, sbuf);
  }
  // 4-5) segment softmax stats
  {
    const long long n = (long long)G_ * ET_;
    k_edge_max<<<(unsigned)((n + 255) / 256), 256, 0, stream>>>(adj, asrc, adst, mbuf);
    k_edge_sum<<<(unsigned)((n + 255) / 256), 256, 0, stream>>>(adj, asrc, adst, mbuf, sbuf);
  }
  // 6) weighted aggregation (coalesced per-channel scatter)
  {
    const long long n = (long long)G_ * ET_ * HC_;
    k_edge_agg<<<(unsigned)((n + 255) / 256), 256, 0, stream>>>(adj, asrc, adst, mbuf, sbuf, xh, out);
  }
}
